// WindowedAttentionBlockNd_90366111907896
// MI455X (gfx1250) — matrix operand order repaired and checked by reading, not compiled, not run on hardware
//
#include <hip/hip_runtime.h>
#include <hip/hip_bf16.h>
#include <math.h>

// Static problem config (matches reference)
#define BB     8
#define CC     256
#define HH     128
#define WW2    128
#define KK     8
#define SS     4
#define NHEADS 8
#define EE     256
#define HD     32
#define HIDD   768
#define LL     64
#define NWIN   256

// LDS plan (bytes):
//  s16a : 64x256 f16  (X -> ctx -> x1 f16)          32768
//  s16b : 64x768 f16  (Q|K|V -> MLP hidden)         98304
//  sf32 : 64x256 f32  (attn_out -> x1 f32)          65536
//  sf32b: 64x256 f32  (mlp_out)                     65536
//  s_sc : 64x64  f32  (scores, one head at a time)  16384
//  s_at : 64x64  f16  (attn probs)                   8192
//  s_rpe: 225x8  f32  (RPE bias table)               7200
#define SMEM_BYTES (32768 + 98304 + 65536 + 65536 + 16384 + 8192 + 7200)

// f16 weight cache layout in d_ws (element offsets)
#define OFF_WQ 0
#define OFF_WK 65536
#define OFF_WV 131072
#define OFF_WP 196608
#define OFF_W1 262144
#define OFF_W2 458752
#define WS_F16_ELEMS 655360  // 1.31 MB

#define RPE_ELEMS (225 * NHEADS)

typedef __attribute__((ext_vector_type(16))) _Float16 v16h;
typedef __attribute__((ext_vector_type(8)))  float    v8f;

// Global (addrspace 1) pointer types: avoid flat_load (LOADcnt+DScnt) for
// provably-global traffic; get global_load_b128 tracked by LOADcnt only.
typedef const float    __attribute__((address_space(1)))* gf32;
typedef const _Float16 __attribute__((address_space(1)))* gh16;
typedef float          __attribute__((address_space(1)))* gf32w;

__device__ __forceinline__ v8f wmma_f16(v16h a, v16h b, v8f c) {
  // D = A(16x32 f16) * B(32x16 f16) + C(16x16 f32)
  return __builtin_amdgcn_wmma_f32_16x16x32_f16(false, a, false, b, (short)0, c,
                                                false, false);
}

__device__ __forceinline__ v8f zero8() {
  v8f z = {0.f, 0.f, 0.f, 0.f, 0.f, 0.f, 0.f, 0.f};
  return z;
}

// A fragment: 16x32 f16 tile at (m0,k0) of row-major [rows][ld] LDS matrix.
// lane<16:  row=lane,    K = k0..k0+15
// lane>=16: row=lane-16, K = k0+16..k0+31
__device__ __forceinline__ v16h load_a_f16(const _Float16* base, int ld, int m0,
                                           int k0, int lane) {
  const _Float16* p = base + (m0 + (lane & 15)) * ld + k0 + ((lane >> 4) << 4);
  v16h a;
#pragma unroll
  for (int i = 0; i < 8; ++i) a[i] = p[i];
#pragma unroll
  for (int i = 0; i < 8; ++i) a[8 + i] = p[8 + i];
  return a;
}

// B fragment where B[k][n] = src[(n0+n)*ld + k0+k] (LDS source indexed [n][k])
// lane<16: col=lane, K=k0..k0+15 ; lane>=16: col=lane-16, K=k0+16..k0+31
__device__ __forceinline__ v16h load_b_nk_f16(const _Float16* base, int ld,
                                              int n0, int k0, int lane) {
  const _Float16* p = base + (n0 + (lane & 15)) * ld + k0 + ((lane >> 4) << 4);
  v16h b;
#pragma unroll
  for (int i = 0; i < 16; ++i) b[i] = p[i];
  return b;
}

// B fragment where B[k][n] = src[(k0+k)*ld + n0+n] (LDS source indexed [k][n])
__device__ __forceinline__ v16h load_b_kn_f16(const _Float16* base, int ld,
                                              int n0, int k0, int lane) {
  const _Float16* p = base + (n0 + (lane & 15)) +
                      (int)(k0 + ((lane >> 4) << 4)) * ld;
  v16h b;
#pragma unroll
  for (int i = 0; i < 16; ++i) b[i] = p[i * ld];
  return b;
}

// Weight B fragment from GLOBAL memory. F16W: pre-converted f16 cache in d_ws
// (two contiguous global_load_b128 per lane). Fallback: f32 + convert.
template <bool F16W>
__device__ __forceinline__ v16h load_w(const float* w32, const _Float16* w16,
                                       int ld, int n0, int k0, int lane) {
  const int off = (n0 + (lane & 15)) * ld + k0 + ((lane >> 4) << 4);
  v16h b;
  if constexpr (F16W) {
    gh16 p = (gh16)(w16 + off);
#pragma unroll
    for (int i = 0; i < 16; ++i) b[i] = p[i];
  } else {
    gf32 p = (gf32)(w32 + off);
#pragma unroll
    for (int i = 0; i < 16; ++i) b[i] = (_Float16)p[i];
  }
  return b;
}

// Swin shift-mask region id on shifted-image coordinate v in [0,128)
__device__ __forceinline__ int regid(int v) {
  return (v < HH - KK) ? 0 : ((v < HH - SS) ? 1 : 2);
}

// gelu(x) = 0.5*x*(1+tanh(u)) = x * sigmoid(2u), u = c*(x + 0.044715 x^3)
// One v_exp_f32 + one v_rcp_f32 instead of an inlined tanh polynomial.
__device__ __forceinline__ float gelu_tanh(float x) {
  float u = 0.79788456080286535588f * (x + 0.044715f * x * x * x);
  float e = __expf(-2.0f * u);
  return x * __builtin_amdgcn_rcpf(1.0f + e);
}

// sum/max over the 4 consecutive lanes sharing one row (quad reduction)
__device__ __forceinline__ float quad_sum(float v) {
  v += __shfl_xor(v, 1, 32);
  v += __shfl_xor(v, 2, 32);
  return v;
}
__device__ __forceinline__ float quad_max(float v) {
  v = fmaxf(v, __shfl_xor(v, 1, 32));
  v = fmaxf(v, __shfl_xor(v, 2, 32));
  return v;
}

__global__ void cvt_w_f16(const float* __restrict__ src,
                          _Float16* __restrict__ dst, int n) {
  int i = blockIdx.x * 256 + threadIdx.x;
  if (i < n) {
    gf32 s = (gf32)src;
    dst[i] = (_Float16)s[i];
  }
}

template <bool F16W>
__global__ __launch_bounds__(256)
void swin_block_kernel(const float* __restrict__ query,
                       const float* __restrict__ w_q, const float* __restrict__ b_q,
                       const float* __restrict__ w_k, const float* __restrict__ b_k,
                       const float* __restrict__ w_v, const float* __restrict__ b_v,
                       const float* __restrict__ w_proj, const float* __restrict__ b_proj,
                       const float* __restrict__ gamma_attn, const float* __restrict__ beta_attn,
                       const float* __restrict__ w1, const float* __restrict__ b1,
                       const float* __restrict__ w2, const float* __restrict__ b2,
                       const float* __restrict__ gamma_mlp, const float* __restrict__ beta_mlp,
                       const float* __restrict__ rpe_table,
                       const _Float16* __restrict__ wf,  // f16 weight cache (d_ws)
                       float* __restrict__ out) {
  extern __shared__ __align__(16) char smem[];
  _Float16* s16a  = (_Float16*)smem;               // 64x256 f16
  _Float16* s16b  = s16a + LL * CC;                // 64x768 f16
  float*    sf32  = (float*)(s16b + LL * HIDD);    // 64x256 f32
  float*    sf32b = sf32 + LL * CC;                // 64x256 f32
  float*    s_sc  = sf32b + LL * CC;               // 64x64 f32
  _Float16* s_at  = (_Float16*)(s_sc + LL * LL);   // 64x64 f16
  float*    s_rpe = (float*)(s_at + LL * LL);      // 225x8 f32

  const int tid  = threadIdx.x;
  const int lane = tid & 31;
  const int wid  = tid >> 5;

  const int bidx = blockIdx.x >> 8;          // batch
  const int win  = blockIdx.x & (NWIN - 1);  // window
  const int wh = win >> 4, ww = win & 15;

  gf32  gq   = (gf32)query;
  gf32  grpe = (gf32)rpe_table;
  gf32w gout = (gf32w)out;

  const _Float16* wq16 = wf + OFF_WQ;
  const _Float16* wk16 = wf + OFF_WK;
  const _Float16* wv16 = wf + OFF_WV;
  const _Float16* wp16 = wf + OFF_WP;
  const _Float16* w116 = wf + OFF_W1;
  const _Float16* w216 = wf + OFF_W2;

  // ---- Phase 0: stage RPE table + gather shifted window (f32 -> f16) ----
  for (int idx = tid; idx < RPE_ELEMS; idx += 256) s_rpe[idx] = grpe[idx];
  for (int idx = tid; idx < LL * CC; idx += 256) {
    int t = idx & (LL - 1);
    int c = idx >> 6;
    int i = t >> 3, j = t & 7;
    int y = (wh * KK + i + SS) & (HH - 1);   // roll(query, -S)
    int x = (ww * KK + j + SS) & (WW2 - 1);
    s16a[t * CC + c] = (_Float16)gq[(((size_t)bidx * CC + c) * HH + y) * WW2 + x];
  }
  __syncthreads();

  // ---- Phase 1: Q,K,V = X * W^T + b (f16 out to s16b) ----
  {
    const float*    Wm[3]   = {w_q, w_k, w_v};
    const _Float16* Wm16[3] = {wq16, wk16, wv16};
    gf32 Bv[3] = {(gf32)b_q, (gf32)b_k, (gf32)b_v};
    for (int m3 = 0; m3 < 3; ++m3) {
      _Float16* dst = s16b + m3 * (LL * CC);
      v8f acc[4][2];
#pragma unroll
      for (int m = 0; m < 4; ++m) { acc[m][0] = zero8(); acc[m][1] = zero8(); }
      const int n0 = (wid * 2) * 16, n1 = n0 + 16;
      for (int kt = 0; kt < 8; ++kt) {
        const int k0 = kt * 32;
        if (kt + 2 < 8) {  // prefetch upcoming weight slab into near cache
          if constexpr (F16W)
            __builtin_prefetch(Wm16[m3] + (n0 + (lane & 15)) * CC + k0 + 64, 0, 3);
          else
            __builtin_prefetch(Wm[m3] + (n0 + (lane & 15)) * CC + k0 + 64, 0, 3);
        }
        v16h bb0 = load_w<F16W>(Wm[m3], Wm16[m3], CC, n0, k0, lane);
        v16h bb1 = load_w<F16W>(Wm[m3], Wm16[m3], CC, n1, k0, lane);
#pragma unroll
        for (int m = 0; m < 4; ++m) {
          v16h a = load_a_f16(s16a, CC, m * 16, k0, lane);
          acc[m][0] = wmma_f16(a, bb0, acc[m][0]);
          acc[m][1] = wmma_f16(a, bb1, acc[m][1]);
        }
      }
#pragma unroll
      for (int m = 0; m < 4; ++m) {
#pragma unroll
        for (int n = 0; n < 2; ++n) {
          const int col = (wid * 2 + n) * 16 + (lane & 15);
          const float bias = Bv[m3][col];
          const int row0 = m * 16 + ((lane >> 4) << 3);
#pragma unroll
          for (int r = 0; r < 8; ++r)
            dst[(row0 + r) * CC + col] = (_Float16)(acc[m][n][r] + bias);
        }
      }
    }
  }
  __syncthreads();

  // ---- Phase 2: per-head attention ----
  const _Float16* s_q = s16b;
  const _Float16* s_k = s16b + LL * CC;
  const _Float16* s_v = s16b + 2 * LL * CC;
  const float scale = 0.17677669529663687f;  // 1/sqrt(32)

  for (int h = 0; h < NHEADS; ++h) {
    // scores = Q_h * K_h^T : 16 tiles -> 2 per wave
#pragma unroll
    for (int u = 0; u < 2; ++u) {
      const int tt = wid * 2 + u;
      const int mt = tt >> 2, nt = tt & 3;
      v16h a  = load_a_f16(s_q, CC, mt * 16, h * HD, lane);
      v16h bb = load_b_nk_f16(s_k, CC, nt * 16, h * HD, lane);
      v8f c = zero8();
      c = wmma_f16(a, bb, c);
      const int col = nt * 16 + (lane & 15);
      const int row0 = mt * 16 + ((lane >> 4) << 3);
#pragma unroll
      for (int r = 0; r < 8; ++r) s_sc[(row0 + r) * LL + col] = c[r];
    }
    __syncthreads();

    // softmax with scale + RPE bias (from LDS) + shift mask.
    // All 256 threads: row y = tid>>2, 4 lanes per row each own 16 cols,
    // quad reductions via shfl_xor; values stay in registers.
    {
      const int y = tid >> 2;
      const int g = tid & 3;
      const int qi = y >> 3, qj = y & 7;
      const int qrid = regid(wh * KK + qi) * 3 + regid(ww * KK + qj);
      float v[16];
      float mx = -3.0e38f;
#pragma unroll
      for (int kk = 0; kk < 16; ++kk) {
        const int k = g * 16 + kk;
        const int ki = k >> 3, kj = k & 7;
        const int krid = regid(wh * KK + ki) * 3 + regid(ww * KK + kj);
        float val;
        if (krid != qrid) {
          val = -1.0e9f;
        } else {
          const int ridx = (qi - ki + 7) * 15 + (qj - kj + 7);
          val = s_sc[y * LL + k] * scale + s_rpe[ridx * NHEADS + h];
        }
        v[kk] = val;
        mx = fmaxf(mx, val);
      }
      mx = quad_max(mx);
      float sum = 0.f;
#pragma unroll
      for (int kk = 0; kk < 16; ++kk) {
        v[kk] = __expf(v[kk] - mx);
        sum += v[kk];
      }
      sum = quad_sum(sum);
      const float inv = 1.0f / sum;
#pragma unroll
      for (int kk = 0; kk < 16; ++kk)
        s_at[y * LL + g * 16 + kk] = (_Float16)(v[kk] * inv);
    }
    __syncthreads();

    // ctx_h = attn * V_h : 8 tiles -> 1 per wave
    {
      const int mt = wid >> 1, nt = wid & 1;
      v8f c = zero8();
#pragma unroll
      for (int kt = 0; kt < 2; ++kt) {
        v16h a  = load_a_f16(s_at, LL, mt * 16, kt * 32, lane);
        v16h bb = load_b_kn_f16(s_v, CC, h * HD + nt * 16, kt * 32, lane);
        c = wmma_f16(a, bb, c);
      }
      const int col = h * HD + nt * 16 + (lane & 15);
      const int row0 = mt * 16 + ((lane >> 4) << 3);
#pragma unroll
      for (int r = 0; r < 8; ++r) s16a[(row0 + r) * CC + col] = (_Float16)c[r];
    }
    __syncthreads();
  }

  // ---- Phase 3: projection: attn_out = ctx * Wproj^T + b -> sf32 ----
  {
    v8f acc[4][2];
#pragma unroll
    for (int m = 0; m < 4; ++m) { acc[m][0] = zero8(); acc[m][1] = zero8(); }
    const int n0 = (wid * 2) * 16, n1 = n0 + 16;
    for (int kt = 0; kt < 8; ++kt) {
      const int k0 = kt * 32;
      v16h bb0 = load_w<F16W>(w_proj, wp16, EE, n0, k0, lane);
      v16h bb1 = load_w<F16W>(w_proj, wp16, EE, n1, k0, lane);
#pragma unroll
      for (int m = 0; m < 4; ++m) {
        v16h a = load_a_f16(s16a, CC, m * 16, k0, lane);
        acc[m][0] = wmma_f16(a, bb0, acc[m][0]);
        acc[m][1] = wmma_f16(a, bb1, acc[m][1]);
      }
    }
    gf32 gbp = (gf32)b_proj;
#pragma unroll
    for (int m = 0; m < 4; ++m) {
#pragma unroll
      for (int n = 0; n < 2; ++n) {
        const int col = (wid * 2 + n) * 16 + (lane & 15);
        const float bias = gbp[col];
        const int row0 = m * 16 + ((lane >> 4) << 3);
#pragma unroll
        for (int r = 0; r < 8; ++r)
          sf32[(row0 + r) * CC + col] = acc[m][n][r] + bias;
      }
    }
  }
  __syncthreads();

  // ---- LayerNorm(attn_out) + residual (re-read shifted input) ----
  // All 256 threads: 4 lanes per row, 64 cols each, quad reductions.
  {
    const int t = tid >> 2;
    const int g = tid & 3;
    const int i = t >> 3, j = t & 7;
    const int y = (wh * KK + i + SS) & (HH - 1);
    const int x = (ww * KK + j + SS) & (WW2 - 1);
    gf32 gga = (gf32)gamma_attn;
    gf32 gba = (gf32)beta_attn;
    float s = 0.f;
    for (int c0 = 0; c0 < 64; ++c0) s += sf32[t * CC + g * 64 + c0];
    const float mean = quad_sum(s) * (1.0f / CC);
    float vs = 0.f;
    for (int c0 = 0; c0 < 64; ++c0) {
      float d = sf32[t * CC + g * 64 + c0] - mean;
      vs += d * d;
    }
    const float rs = rsqrtf(quad_sum(vs) * (1.0f / CC) + 1e-5f);
    for (int c0 = 0; c0 < 64; ++c0) {
      const int c = g * 64 + c0;
      const float xw = gq[(((size_t)bidx * CC + c) * HH + y) * WW2 + x];
      const float x1 = (sf32[t * CC + c] - mean) * rs * gga[c] + gba[c] + xw;
      sf32[t * CC + c] = x1;             // f32 for final residual
      s16a[t * CC + c] = (_Float16)x1;   // f16 input to MLP
    }
  }
  __syncthreads();

  // ---- Phase 4: MLP1: h = gelu(x1 * W1^T + b1) -> s16b (64x768 f16) ----
  {
    gf32 gb1 = (gf32)b1;
    for (int z = 0; z < 6; ++z) {
      const int n0 = (wid + z * 8) * 16;
      v8f acc[4];
#pragma unroll
      for (int m = 0; m < 4; ++m) acc[m] = zero8();
      for (int kt = 0; kt < 8; ++kt) {
        const int k0 = kt * 32;
        v16h bb = load_w<F16W>(w1, w116, CC, n0, k0, lane);
#pragma unroll
        for (int m = 0; m < 4; ++m) {
          v16h a = load_a_f16(s16a, CC, m * 16, k0, lane);
          acc[m] = wmma_f16(a, bb, acc[m]);
        }
      }
      const int col = n0 + (lane & 15);
      const float bias = gb1[col];
#pragma unroll
      for (int m = 0; m < 4; ++m) {
        const int row0 = m * 16 + ((lane >> 4) << 3);
#pragma unroll
        for (int r = 0; r < 8; ++r)
          s16b[(row0 + r) * HIDD + col] = (_Float16)gelu_tanh(acc[m][r] + bias);
      }
    }
  }
  __syncthreads();

  // ---- Phase 5: MLP2: mlp_out = h * W2^T + b2 -> sf32b ----
  {
    v8f acc[4][2];
#pragma unroll
    for (int m = 0; m < 4; ++m) { acc[m][0] = zero8(); acc[m][1] = zero8(); }
    const int n0 = (wid * 2) * 16, n1 = n0 + 16;
    for (int kt = 0; kt < 24; ++kt) {
      const int k0 = kt * 32;
      if (kt + 2 < 24) {
        if constexpr (F16W)
          __builtin_prefetch(w216 + (n0 + (lane & 15)) * HIDD + k0 + 64, 0, 3);
        else
          __builtin_prefetch(w2 + (n0 + (lane & 15)) * HIDD + k0 + 64, 0, 3);
      }
      v16h bb0 = load_w<F16W>(w2, w216, HIDD, n0, k0, lane);
      v16h bb1 = load_w<F16W>(w2, w216, HIDD, n1, k0, lane);
#pragma unroll
      for (int m = 0; m < 4; ++m) {
        v16h a = load_a_f16(s16b, HIDD, m * 16, k0, lane);
        acc[m][0] = wmma_f16(a, bb0, acc[m][0]);
        acc[m][1] = wmma_f16(a, bb1, acc[m][1]);
      }
    }
    gf32 gb2 = (gf32)b2;
#pragma unroll
    for (int m = 0; m < 4; ++m) {
#pragma unroll
      for (int n = 0; n < 2; ++n) {
        const int col = (wid * 2 + n) * 16 + (lane & 15);
        const float bias = gb2[col];
        const int row0 = m * 16 + ((lane >> 4) << 3);
#pragma unroll
        for (int r = 0; r < 8; ++r)
          sf32b[(row0 + r) * CC + col] = acc[m][n][r] + bias;
      }
    }
  }
  __syncthreads();

  // ---- Final: LayerNorm(mlp_out) + x1, scatter back (un-shift) ----
  {
    const int t = tid >> 2;
    const int g = tid & 3;
    const int i = t >> 3, j = t & 7;
    const int y = (wh * KK + i + SS) & (HH - 1);
    const int x = (ww * KK + j + SS) & (WW2 - 1);
    gf32 ggm = (gf32)gamma_mlp;
    gf32 gbm = (gf32)beta_mlp;
    float s = 0.f;
    for (int c0 = 0; c0 < 64; ++c0) s += sf32b[t * CC + g * 64 + c0];
    const float mean = quad_sum(s) * (1.0f / CC);
    float vs = 0.f;
    for (int c0 = 0; c0 < 64; ++c0) {
      float d = sf32b[t * CC + g * 64 + c0] - mean;
      vs += d * d;
    }
    const float rs = rsqrtf(quad_sum(vs) * (1.0f / CC) + 1e-5f);
    for (int c0 = 0; c0 < 64; ++c0) {
      const int c = g * 64 + c0;
      const float val =
          (sf32b[t * CC + c] - mean) * rs * ggm[c] + gbm[c] + sf32[t * CC + c];
      gout[(((size_t)bidx * CC + c) * HH + y) * WW2 + x] = val;
    }
  }
}

extern "C" void kernel_launch(void* const* d_in, const int* in_sizes, int n_in,
                              void* d_out, int out_size, void* d_ws, size_t ws_size,
                              hipStream_t stream) {
  (void)in_sizes; (void)n_in; (void)out_size;
  const float* query      = (const float*)d_in[0];
  const float* w_q        = (const float*)d_in[1];
  const float* b_q        = (const float*)d_in[2];
  const float* w_k        = (const float*)d_in[3];
  const float* b_k        = (const float*)d_in[4];
  const float* w_v        = (const float*)d_in[5];
  const float* b_v        = (const float*)d_in[6];
  const float* w_proj     = (const float*)d_in[7];
  const float* b_proj     = (const float*)d_in[8];
  const float* gamma_attn = (const float*)d_in[9];
  const float* beta_attn  = (const float*)d_in[10];
  const float* w1         = (const float*)d_in[11];
  const float* b1         = (const float*)d_in[12];
  const float* w2         = (const float*)d_in[13];
  const float* b2         = (const float*)d_in[14];
  const float* gamma_mlp  = (const float*)d_in[15];
  const float* beta_mlp   = (const float*)d_in[16];
  const float* rpe        = (const float*)d_in[17];
  float* out = (float*)d_out;

  const bool f16w = ws_size >= (size_t)WS_F16_ELEMS * sizeof(_Float16);
  _Float16* wf = (_Float16*)d_ws;

  if (f16w) {
    // Pre-convert weights to f16 once per launch (1.31 MB, L2 resident):
    // halves per-block weight read traffic and removes cvt from GEMM loops.
    cvt_w_f16<<<dim3(EE * CC / 256), dim3(256), 0, stream>>>(w_q, wf + OFF_WQ, EE * CC);
    cvt_w_f16<<<dim3(EE * CC / 256), dim3(256), 0, stream>>>(w_k, wf + OFF_WK, EE * CC);
    cvt_w_f16<<<dim3(EE * CC / 256), dim3(256), 0, stream>>>(w_v, wf + OFF_WV, EE * CC);
    cvt_w_f16<<<dim3(CC * EE / 256), dim3(256), 0, stream>>>(w_proj, wf + OFF_WP, CC * EE);
    cvt_w_f16<<<dim3(HIDD * CC / 256), dim3(256), 0, stream>>>(w1, wf + OFF_W1, HIDD * CC);
    cvt_w_f16<<<dim3(CC * HIDD / 256), dim3(256), 0, stream>>>(w2, wf + OFF_W2, CC * HIDD);

    (void)hipFuncSetAttribute(
        reinterpret_cast<const void*>(&swin_block_kernel<true>),
        hipFuncAttributeMaxDynamicSharedMemorySize, SMEM_BYTES);
    swin_block_kernel<true><<<dim3(BB * NWIN), dim3(256), SMEM_BYTES, stream>>>(
        query, w_q, b_q, w_k, b_k, w_v, b_v, w_proj, b_proj, gamma_attn,
        beta_attn, w1, b1, w2, b2, gamma_mlp, beta_mlp, rpe, wf, out);
  } else {
    (void)hipFuncSetAttribute(
        reinterpret_cast<const void*>(&swin_block_kernel<false>),
        hipFuncAttributeMaxDynamicSharedMemorySize, SMEM_BYTES);
    swin_block_kernel<false><<<dim3(BB * NWIN), dim3(256), SMEM_BYTES, stream>>>(
        query, w_q, b_q, w_k, b_k, w_v, b_v, w_proj, b_proj, gamma_attn,
        beta_attn, w1, b1, w2, b2, gamma_mlp, beta_mlp, rpe, wf, out);
  }
}
